// Informer_65343632442055
// MI455X (gfx1250) — matrix operand order repaired and checked by reading, not compiled, not run on hardware
//
#include <hip/hip_runtime.h>
#include <cmath>

// ---------------------------------------------------------------------------
// CDNA5 (gfx1250) types for WMMA
// ---------------------------------------------------------------------------
typedef __attribute__((ext_vector_type(16))) __bf16         v16bf;
typedef __attribute__((ext_vector_type(8)))  float          v8f;
typedef __attribute__((ext_vector_type(8)))  unsigned short v8us;
typedef __attribute__((ext_vector_type(16))) unsigned short v16us;

__device__ __forceinline__ unsigned short f2bf(float f) {
  unsigned int u = __float_as_uint(f);
  u += 0x7FFFu + ((u >> 16) & 1u);   // round-to-nearest-even
  return (unsigned short)(u >> 16);
}

// ---------------------------------------------------------------------------
// CDNA5 async global->LDS copy (one hop, tracked by ASYNCcnt), inline asm so
// it is portable across ROCm 7.2 / amdgpu-toolchain builtin arity differences.
// lds_off = LDS byte offset (low 32 bits of flat shared pointer).
// ---------------------------------------------------------------------------
__device__ __forceinline__ void inf_async_ld_b128(unsigned lds_off,
                                                  const unsigned short* g) {
  asm volatile("global_load_async_to_lds_b128 %0, %1, off"
               :: "v"(lds_off), "v"(g) : "memory");
}
__device__ __forceinline__ void inf_wait_async_le1() {
  asm volatile("s_wait_asynccnt 0x1" ::: "memory");
}
__device__ __forceinline__ void inf_wait_async_0() {
  asm volatile("s_wait_asynccnt 0x0" ::: "memory");
}

// ---------------------------------------------------------------------------
// WMMA GEMM:  C[M,N] = A[M,K](bf16) * W[N,K]^T(bf16) (+bias) (+=C if accum)
// Row remap for batched wrap-padded inputs:
//   aRow = (r / rowsC_pb) * rowsA_pb + (r % rowsC_pb) + rowOff
// Block: 256 threads = 8 waves; each wave -> 16x64 tile (4 x 16x16 WMMA tiles)
// B tile (32K x 64N) double-buffered in LDS via async loads: tile k+1 is in
// flight while tile k feeds the matrix pipes.
// ---------------------------------------------------------------------------
__global__ __launch_bounds__(256)
void inf_wmma_gemm_bf16(const unsigned short* __restrict__ A,
                        const unsigned short* __restrict__ W,
                        const float* __restrict__ bias,
                        float* __restrict__ C,
                        int M, int N, int K,
                        int rowsC_pb, int rowsA_pb, int rowOff, int accum)
{
  __shared__ unsigned short Bs[2][64 * 32];
  const int tid  = threadIdx.x;
  const int lane = tid & 31;          // wave32
  const int wave = tid >> 5;
  const int lm   = lane & 15;
  const int hs   = lane >> 4;         // half-wave select
  const int nBase = blockIdx.x * 64;
  const int mBase = blockIdx.y * 128 + wave * 16;

  // A row for this lane (CDNA5 A-frag: lane 0..15 holds row M=lane%16)
  int mRow = mBase + lm;
  if (mRow > M - 1) mRow = M - 1;
  const int bb = mRow / rowsC_pb;
  const size_t aRow = (size_t)bb * rowsA_pb + (size_t)(mRow - bb * rowsC_pb) + rowOff;
  const unsigned short* Ap = A + aRow * (size_t)K;

  union AF { struct { v8us lo, hi; } p; v16bf v; };
  union BF { v16us u; v16bf v; };

  v8f acc[4] = {};
  if (accum) {
#pragma unroll
    for (int t = 0; t < 4; ++t)
#pragma unroll
      for (int r = 0; r < 8; ++r) {
        int row = mBase + r + hs * 8;
        int col = nBase + t * 16 + lm;
        if (row < M) acc[t][r] = C[(size_t)row * N + col];
      }
  }

  // cooperative async-fill assignment: 256 threads x 16B = 64x32 bf16 tile
  const int fillN = tid >> 2;
  const int fillK = (tid & 3) * 8;
  const unsigned short* Wfill = W + (size_t)(nBase + fillN) * K + fillK;
  const unsigned ldsDst[2] = {
    (unsigned)(unsigned long long)(const void*)&Bs[0][fillN * 32 + fillK],
    (unsigned)(unsigned long long)(const void*)&Bs[1][fillN * 32 + fillK]
  };

  const int nk = K >> 5;
  inf_async_ld_b128(ldsDst[0], Wfill);            // preload tile 0

  for (int ik = 0; ik < nk; ++ik) {
    const int kk = ik << 5;
    if (ik + 1 < nk) {
      inf_async_ld_b128(ldsDst[(ik + 1) & 1], Wfill + kk + 32);  // overlap next tile
      inf_wait_async_le1();                       // oldest (tile ik) complete
    } else {
      inf_wait_async_0();
    }
    __syncthreads();                              // tile ik resident for all waves

    if (kk + 64 < K) __builtin_prefetch(Ap + kk + 64, 0, 3);  // near-scope prefetch

    // A fragment 16x32 (M x K), contiguous K order used here (B matches):
    //  lanes 0-15: K {0..15}; lanes 16-31: K {16..31}
    AF a;
    a.p.lo = *(const v8us*)(Ap + kk + hs * 16);
    a.p.hi = *(const v8us*)(Ap + kk + hs * 16 + 8);

    const unsigned short* Bt = Bs[ik & 1];
#pragma unroll
    for (int t = 0; t < 4; ++t) {
      // B fragment 32x16 (K x N): lane = N column, 16 contiguous K per lane
      BF b;
      b.u = *(const v16us*)(&Bt[(t * 16 + lm) * 32 + hs * 16]);
      acc[t] = __builtin_amdgcn_wmma_f32_16x16x32_bf16(
          false, a.v, false, b.v, (short)0, acc[t], false, false);
    }
    __syncthreads();                              // reads done before buffer reuse
  }

#pragma unroll
  for (int t = 0; t < 4; ++t)
#pragma unroll
    for (int r = 0; r < 8; ++r) {
      int row = mBase + r + hs * 8;
      if (row < M) {
        int col = nBase + t * 16 + lm;
        float v = acc[t][r];
        if (bias) v += bias[col];
        C[(size_t)row * N + col] = v;
      }
    }
}

// ---------------------------------------------------------------------------
// Elementwise / small kernels
// ---------------------------------------------------------------------------
__global__ void inf_cvt_bf16(const float* __restrict__ in,
                             unsigned short* __restrict__ out, long n) {
  long i = (long)blockIdx.x * 256 + threadIdx.x;
  if (i < n) out[i] = f2bf(in[i]);
}

// dcw (lay, dout, din, 3) -> bf16 (lay, kw, dout, din)
__global__ void inf_cvt_dcw(const float* __restrict__ dcw,
                            unsigned short* __restrict__ out, long n) {
  long i = (long)blockIdx.x * 256 + threadIdx.x;
  if (i >= n) return;
  int din  = (int)(i % 512);
  int dout = (int)((i / 512) % 512);
  int kw   = (int)((i / (512 * 512)) % 3);
  int lay  = (int)(i / (3L * 512 * 512));
  out[i] = f2bf(dcw[(((size_t)lay * 512 + dout) * 512 + din) * 3 + kw]);
}

// circular embedding conv: C=7 -> Dm=512, width 3
__global__ void inf_embed(const float* __restrict__ xe, const float* __restrict__ ew,
                          float* __restrict__ x, unsigned short* __restrict__ xb,
                          int B, int L) {
  long i = (long)blockIdx.x * 256 + threadIdx.x;
  long n = (long)B * L * 512;
  if (i >= n) return;
  int d = (int)(i % 512);
  int l = (int)((i / 512) % L);
  int b = (int)(i / (512L * L));
  float acc = 0.f;
#pragma unroll
  for (int t = 0; t < 3; ++t) {
    int ls = (l + t - 1 + L) % L;
    const float* xr = xe + ((size_t)b * L + ls) * 7;
    const float* w  = ew + (size_t)d * 21 + t;  // ew[d][c][t]
#pragma unroll
    for (int c = 0; c < 7; ++c) acc += xr[c] * w[c * 3];
  }
  x[i] = acc;
  xb[i] = f2bf(acc);
}

// ProbSparse M-score with deterministic hash sampling (replaces jax PRNG)
__global__ void inf_mscore(const float* __restrict__ Qf, const float* __restrict__ Kf,
                           float* __restrict__ Msc, int B, int H, int L, int u, int layer) {
  long i = (long)blockIdx.x * 256 + threadIdx.x;
  long n = (long)B * H * L;
  if (i >= n) return;
  int l = (int)(i % L);
  int h = (int)((i / L) % H);
  int b = (int)(i / ((long)L * H));
  const float* q = Qf + ((size_t)b * L + l) * 512 + h * 64;
  float mx = -1e30f, sm = 0.f;
  for (int j = 0; j < u; ++j) {
    unsigned s = (unsigned)l * 2654435761u ^ ((unsigned)j * 0x9E3779B9u) ^ ((unsigned)layer * 0x85EBCA6Bu);
    s ^= s >> 16; s *= 0x7FEB352Du; s ^= s >> 15; s *= 0x846CA68Bu; s ^= s >> 16;
    int kl = (int)(s % (unsigned)L);
    const float* kp = Kf + ((size_t)b * L + kl) * 512 + h * 64;
    float d = 0.f;
#pragma unroll 8
    for (int t = 0; t < 64; ++t) d += q[t] * kp[t];
    mx = fmaxf(mx, d);
    sm += d;
  }
  Msc[((size_t)b * H + h) * L + l] = mx - sm / (float)L;
}

// per-(b,h) top-u selection (u <= 24): iterative argmax, destroys Msc (scratch)
__global__ __launch_bounds__(256)
void inf_topk(float* __restrict__ Msc, int* __restrict__ out, int L, int u) {
  __shared__ float rv[256];
  __shared__ int   ri[256];
  const int tid = threadIdx.x;
  float* Mr = Msc + (size_t)blockIdx.x * L;
  for (int t = 0; t < u; ++t) {
    float bv = -1e30f; int bi = 0;
    for (int l = tid; l < L; l += 256)
      if (Mr[l] > bv) { bv = Mr[l]; bi = l; }
    rv[tid] = bv; ri[tid] = bi;
    __syncthreads();
    for (int o = 128; o > 0; o >>= 1) {
      if (tid < o && rv[tid + o] > rv[tid]) { rv[tid] = rv[tid + o]; ri[tid] = ri[tid + o]; }
      __syncthreads();
    }
    if (tid == 0) { out[(size_t)blockIdx.x * u + t] = ri[0]; Mr[ri[0]] = -1e30f; }
    __syncthreads();
  }
}

__global__ void inf_vmean(const float* __restrict__ Vf, float* __restrict__ vm,
                          int B, int H, int L) {
  int i = blockIdx.x * 256 + threadIdx.x;
  if (i >= B * H * 64) return;
  int d = i & 63, h = (i >> 6) % H, b = i / (64 * H);
  float s = 0.f;
  for (int l = 0; l < L; ++l) s += Vf[((size_t)b * L + l) * 512 + h * 64 + d];
  vm[i] = s / (float)L;
}

__global__ void inf_fillctx(const float* __restrict__ vm, float* __restrict__ ctx,
                            int B, int H, int L) {
  long i = (long)blockIdx.x * 256 + threadIdx.x;
  long n = (long)B * L * 512;
  if (i >= n) return;
  int dd = (int)(i % 512);
  int b  = (int)(i / (512L * L));
  ctx[i] = vm[((size_t)b * H + (dd >> 6)) * 64 + (dd & 63)];
}

// full softmax attention for one selected query; writes its ctx row
__global__ __launch_bounds__(256)
void inf_attn_top(const float* __restrict__ Qf, const float* __restrict__ Kf,
                  const float* __restrict__ Vf, const int* __restrict__ topi,
                  float* __restrict__ ctx, int B, int H, int L, int u) {
  __shared__ float sc[2048];
  __shared__ float red[256];
  __shared__ float sq[64];
  const int tid = threadIdx.x;
  const int bh = blockIdx.y;
  const int b = bh / H, h = bh % H;
  const int m = topi[(size_t)bh * u + blockIdx.x];
  if (tid < 64) sq[tid] = Qf[((size_t)b * L + m) * 512 + h * 64 + tid];
  __syncthreads();
  float lmax = -1e30f;
  for (int l = tid; l < L; l += 256) {
    const float* kp = Kf + ((size_t)b * L + l) * 512 + h * 64;
    float s = 0.f;
#pragma unroll 8
    for (int d = 0; d < 64; ++d) s += sq[d] * kp[d];
    s *= 0.125f;                       // 1/sqrt(64)
    sc[l] = s;
    lmax = fmaxf(lmax, s);
  }
  red[tid] = lmax; __syncthreads();
  for (int o = 128; o > 0; o >>= 1) { if (tid < o) red[tid] = fmaxf(red[tid], red[tid + o]); __syncthreads(); }
  float smax = red[0]; __syncthreads();
  float lsum = 0.f;
  for (int l = tid; l < L; l += 256) { float p = expf(sc[l] - smax); sc[l] = p; lsum += p; }
  red[tid] = lsum; __syncthreads();
  for (int o = 128; o > 0; o >>= 1) { if (tid < o) red[tid] += red[tid + o]; __syncthreads(); }
  float ssum = red[0]; __syncthreads();
  const int d = tid & 63, grp = tid >> 6;
  float acc = 0.f;
  for (int l = grp; l < L; l += 4) acc += sc[l] * Vf[((size_t)b * L + l) * 512 + h * 64 + d];
  red[tid] = acc; __syncthreads();
  if (grp == 0) {
    float tot = red[d] + red[64 + d] + red[128 + d] + red[192 + d];
    ctx[((size_t)b * L + m) * 512 + h * 64 + d] = tot / ssum;
  }
}

// LayerNorm over D, optional residual input; writes f32 + bf16
__global__ __launch_bounds__(256)
void inf_layernorm(const float* __restrict__ in, const float* __restrict__ res,
                   const float* __restrict__ g, const float* __restrict__ bta,
                   float* __restrict__ out, unsigned short* __restrict__ outb, int D) {
  __shared__ float red[256];
  const int tid = threadIdx.x;
  const size_t row = blockIdx.x;
  const float* xr = in + row * D;
  const float* rr = res ? res + row * D : nullptr;
  float vals[8];
  float s = 0.f;
  int nv = 0;
  for (int i = tid; i < D; i += 256, ++nv) {
    float v = xr[i] + (rr ? rr[i] : 0.f);
    vals[nv] = v; s += v;
  }
  red[tid] = s; __syncthreads();
  for (int o = 128; o > 0; o >>= 1) { if (tid < o) red[tid] += red[tid + o]; __syncthreads(); }
  float mean = red[0] / D; __syncthreads();
  float vs = 0.f;
  for (int j = 0; j < nv; ++j) { float dv = vals[j] - mean; vs += dv * dv; }
  red[tid] = vs; __syncthreads();
  for (int o = 128; o > 0; o >>= 1) { if (tid < o) red[tid] += red[tid + o]; __syncthreads(); }
  float rstd = rsqrtf(red[0] / D + 1e-5f);
  int i = tid;
  for (int j = 0; j < nv; ++j, i += 256) {
    float v = (vals[j] - mean) * rstd * g[i] + bta[i];
    out[row * D + i] = v;
    if (outb) outb[row * D + i] = f2bf(v);
  }
}

__global__ void inf_gelu(float* __restrict__ y, long n) {
  long i = (long)blockIdx.x * 256 + threadIdx.x;
  if (i >= n) return;
  float v = y[i];
  y[i] = 0.5f * v * (1.f + erff(v * 0.70710678118f));
}

// wrap-pad bf16 activations along L: xpad[b][l'] = x[b][(l'-1+L)%L]
__global__ void inf_padwrap(const unsigned short* __restrict__ xb,
                            unsigned short* __restrict__ xp, int B, int L) {
  long i = (long)blockIdx.x * 256 + threadIdx.x;
  long n = (long)B * (L + 2) * 512;
  if (i >= n) return;
  int d = (int)(i % 512);
  long r = i / 512;
  int lp = (int)(r % (L + 2));
  int b  = (int)(r / (L + 2));
  int ls = (lp - 1 + L) % L;
  xp[i] = xb[((size_t)b * L + ls) * 512 + d];
}

// distil epilogue: per-channel scale + ELU + maxpool(win 3, stride 2, pad 1)
__global__ void inf_distil_post(const float* __restrict__ y, const float* __restrict__ bng,
                                const float* __restrict__ bnb, float* __restrict__ x,
                                unsigned short* __restrict__ xb, int B, int L) {
  const int L2 = L / 2;
  long i = (long)blockIdx.x * 256 + threadIdx.x;
  long n = (long)B * L2 * 512;
  if (i >= n) return;
  int d  = (int)(i % 512);
  int lo = (int)((i / 512) % L2);
  int b  = (int)(i / (512L * L2));
  const float scale = bng[d] * rsqrtf(1.f + 1e-5f);
  const float be = bnb[d];
  float m = -1e30f;
#pragma unroll
  for (int t = -1; t <= 1; ++t) {
    int li = 2 * lo + t;
    if (li < 0 || li >= L) continue;
    float v = y[((size_t)b * L + li) * 512 + d] * scale + be;
    v = v > 0.f ? v : (expf(v) - 1.f);   // ELU
    m = fmaxf(m, v);
  }
  size_t o = ((size_t)b * L2 + lo) * 512 + d;
  x[o] = m;
  xb[o] = f2bf(m);
}

__global__ void inf_meanL(const float* __restrict__ x, float* __restrict__ out,
                          int B, int L, int D) {
  int i = blockIdx.x * 256 + threadIdx.x;
  if (i >= B * D) return;
  int d = i % D, b = i / D;
  float s = 0.f;
  for (int l = 0; l < L; ++l) s += x[((size_t)b * L + l) * D + d];
  out[i] = s / (float)L;
}

// ---------------------------------------------------------------------------
// Host orchestration
// ---------------------------------------------------------------------------
extern "C" void kernel_launch(void* const* d_in, const int* in_sizes, int n_in,
                              void* d_out, int out_size, void* d_ws, size_t ws_size,
                              hipStream_t stream) {
  const int B = 32, H = 8, Dm = 512, NL = 3;
  const float* x_enc = (const float*)d_in[0];
  const float* emb_w = (const float*)d_in[1];
  const float* wq = (const float*)d_in[2];   const float* bq = (const float*)d_in[3];
  const float* wk = (const float*)d_in[4];   const float* bk = (const float*)d_in[5];
  const float* wv = (const float*)d_in[6];   const float* bv = (const float*)d_in[7];
  const float* wo = (const float*)d_in[8];   const float* bo = (const float*)d_in[9];
  const float* c1w = (const float*)d_in[10]; const float* c1b = (const float*)d_in[11];
  const float* c2w = (const float*)d_in[12]; const float* c2b = (const float*)d_in[13];
  const float* ln1g = (const float*)d_in[14]; const float* ln1b = (const float*)d_in[15];
  const float* ln2g = (const float*)d_in[16]; const float* ln2b = (const float*)d_in[17];
  const float* dcw = (const float*)d_in[18]; const float* dcb = (const float*)d_in[19];
  const float* bng = (const float*)d_in[20]; const float* bnb = (const float*)d_in[21];
  const float* lnfg = (const float*)d_in[22]; const float* lnfb = (const float*)d_in[23];

  char* base = (char*)d_ws;
  size_t off = 0;
  auto alloc = [&](size_t bytes) -> void* {
    void* p = base + off;
    off = (off + bytes + 255) & ~(size_t)255;
    return p;
  };
  const size_t MR = (size_t)B * 2048;
  float* X            = (float*)alloc(MR * Dm * 4);
  unsigned short* XB  = (unsigned short*)alloc(MR * Dm * 2);
  float* Qf           = (float*)alloc(MR * Dm * 4);   // also reused as bf16 pad buffer
  float* Kf           = (float*)alloc(MR * Dm * 4);
  float* Vf           = (float*)alloc(MR * Dm * 4);
  float* TMP          = (float*)alloc(MR * Dm * 4);
  unsigned short* TMPB = (unsigned short*)alloc(MR * Dm * 2);
  float* SCb          = (float*)alloc((size_t)B * H * 2048 * 4);
  int* TOPI           = (int*)alloc((size_t)B * H * 32 * 4);
  float* VM           = (float*)alloc((size_t)B * H * 64 * 4);
  const size_t WSZ = (size_t)NL * Dm * Dm;
  unsigned short* wqB = (unsigned short*)alloc(WSZ * 2);
  unsigned short* wkB = (unsigned short*)alloc(WSZ * 2);
  unsigned short* wvB = (unsigned short*)alloc(WSZ * 2);
  unsigned short* woB = (unsigned short*)alloc(WSZ * 2);
  unsigned short* c1B = (unsigned short*)alloc(WSZ * 2);
  unsigned short* c2B = (unsigned short*)alloc(WSZ * 2);
  unsigned short* dcB = (unsigned short*)alloc((size_t)2 * 3 * Dm * Dm * 2);
  unsigned short* XPAD = (unsigned short*)Qf;   // lifetime-disjoint reuse

  auto nb = [](long n) { return dim3((unsigned)((n + 255) / 256)); };

  // ---- weight conversion (recomputed every call: deterministic) ----
  inf_cvt_bf16<<<nb(WSZ), 256, 0, stream>>>(wq, wqB, (long)WSZ);
  inf_cvt_bf16<<<nb(WSZ), 256, 0, stream>>>(wk, wkB, (long)WSZ);
  inf_cvt_bf16<<<nb(WSZ), 256, 0, stream>>>(wv, wvB, (long)WSZ);
  inf_cvt_bf16<<<nb(WSZ), 256, 0, stream>>>(wo, woB, (long)WSZ);
  inf_cvt_bf16<<<nb(WSZ), 256, 0, stream>>>(c1w, c1B, (long)WSZ);
  inf_cvt_bf16<<<nb(WSZ), 256, 0, stream>>>(c2w, c2B, (long)WSZ);
  inf_cvt_dcw<<<nb(2L * 3 * Dm * Dm), 256, 0, stream>>>(dcw, dcB, 2L * 3 * Dm * Dm);

  // ---- embedding conv ----
  inf_embed<<<nb((long)B * 2048 * Dm), 256, 0, stream>>>(x_enc, emb_w, X, XB, B, 2048);

  int L = 2048;
  for (int i = 0; i < NL; ++i) {
    const int Mrows = B * L;
    int u = 3 * (int)std::ceil(std::log((double)L));
    if (u > L) u = L;
    const long NE = (long)Mrows * Dm;
    dim3 gg(Dm / 64, (Mrows + 127) / 128);

    // Q, K, V projections (bf16 WMMA, f32 accumulate)
    inf_wmma_gemm_bf16<<<gg, 256, 0, stream>>>(XB, wqB + (size_t)i * Dm * Dm, bq + (size_t)i * Dm,
                                               Qf, Mrows, Dm, Dm, L, L, 0, 0);
    inf_wmma_gemm_bf16<<<gg, 256, 0, stream>>>(XB, wkB + (size_t)i * Dm * Dm, bk + (size_t)i * Dm,
                                               Kf, Mrows, Dm, Dm, L, L, 0, 0);
    inf_wmma_gemm_bf16<<<gg, 256, 0, stream>>>(XB, wvB + (size_t)i * Dm * Dm, bv + (size_t)i * Dm,
                                               Vf, Mrows, Dm, Dm, L, L, 0, 0);

    // ProbSparse attention
    inf_mscore<<<nb((long)B * H * L), 256, 0, stream>>>(Qf, Kf, SCb, B, H, L, u, i);
    inf_topk<<<B * H, 256, 0, stream>>>(SCb, TOPI, L, u);
    inf_vmean<<<nb((long)B * H * 64), 256, 0, stream>>>(Vf, VM, B, H, L);
    inf_fillctx<<<nb(NE), 256, 0, stream>>>(VM, TMP, B, H, L);
    inf_attn_top<<<dim3(u, B * H), 256, 0, stream>>>(Qf, Kf, Vf, TOPI, TMP, B, H, L, u);

    // output projection + residual LN
    inf_cvt_bf16<<<nb(NE), 256, 0, stream>>>(TMP, TMPB, NE);
    inf_wmma_gemm_bf16<<<gg, 256, 0, stream>>>(TMPB, woB + (size_t)i * Dm * Dm, bo + (size_t)i * Dm,
                                               TMP, Mrows, Dm, Dm, L, L, 0, 0);
    inf_layernorm<<<Mrows, 256, 0, stream>>>(TMP, X, ln1g + (size_t)i * Dm, ln1b + (size_t)i * Dm,
                                             X, XB, Dm);

    // FFN
    inf_wmma_gemm_bf16<<<gg, 256, 0, stream>>>(XB, c1B + (size_t)i * Dm * Dm, c1b + (size_t)i * Dm,
                                               TMP, Mrows, Dm, Dm, L, L, 0, 0);
    inf_gelu<<<nb(NE), 256, 0, stream>>>(TMP, NE);
    inf_cvt_bf16<<<nb(NE), 256, 0, stream>>>(TMP, TMPB, NE);
    inf_wmma_gemm_bf16<<<gg, 256, 0, stream>>>(TMPB, c2B + (size_t)i * Dm * Dm, c2b + (size_t)i * Dm,
                                               TMP, Mrows, Dm, Dm, L, L, 0, 0);
    inf_layernorm<<<Mrows, 256, 0, stream>>>(TMP, X, ln2g + (size_t)i * Dm, ln2b + (size_t)i * Dm,
                                             X, XB, Dm);

    // distil: wrap-padded Dm x Dm x 3 conv as 3 accumulated WMMA GEMMs
    if (i < NL - 1) {
      inf_padwrap<<<nb((long)B * (L + 2) * Dm), 256, 0, stream>>>(XB, XPAD, B, L);
      for (int kw = 0; kw < 3; ++kw)
        inf_wmma_gemm_bf16<<<gg, 256, 0, stream>>>(
            XPAD, dcB + ((size_t)(i * 3 + kw)) * Dm * Dm,
            kw == 0 ? dcb + (size_t)i * Dm : nullptr,
            TMP, Mrows, Dm, Dm, L, L + 2, kw, kw > 0 ? 1 : 0);
      inf_distil_post<<<nb((long)B * (L / 2) * Dm), 256, 0, stream>>>(
          TMP, bng + (size_t)i * Dm, bnb + (size_t)i * Dm, X, XB, B, L);
      L /= 2;
    }
  }

  // final LN + mean over L
  inf_layernorm<<<B * L, 256, 0, stream>>>(X, nullptr, lnfg, lnfb, X, XB, Dm);
  inf_meanL<<<nb((long)B * Dm), 256, 0, stream>>>(X, (float*)d_out, B, L, Dm);
}
